// PrototypeLayer_20701742367002
// MI455X (gfx1250) — matrix operand order repaired and checked by reading, not compiled, not run on hardware
//
#include <hip/hip_runtime.h>

// ---------------------------------------------------------------------------
// PrototypeLayer: dist[m,k] = ||x_m||^2 + ||p_k||^2 - 2 * <x_m, p_k>
//   X: [131072, 128] fp32 (B*L rows), P: [64, 128] fp32
//   out: [131072, 64] fp32 distances, then [64,128] prototypes (tuple)
//
// Strategy (MI455X / gfx1250, wave32):
//   - Memory-bound problem (~96 MB @ 23.3 TB/s ~= 4.1 us). Sustaining that
//     needs ~520 TFLOPS on the cross-term GEMM -> fp32 WMMA (16x16x4) is too
//     slow; use split-bf16 (hi+lo) with v_wmma_f32_16x16x32_bf16, 3 WMMAs per
//     tile (hi*hi + hi*lo + lo*hi), fp32 accumulation: ~fp32 accuracy at bf16
//     matrix throughput, stays memory-bound.
//   - Per block (256 thr = 8 waves): 128 rows x all 64 prototypes.
//     LDS staging of bf16 hi/lo tiles (padded stride 136 shorts for alignment
//     + bank behavior), ds_load_b128 fragment fetches, contiguous k = 16h + s in A and B.
//   - Staging uses b128 global loads (512 B/wave/instr) + packed b64 LDS
//     stores to minimize instruction pressure on the bandwidth-critical path.
// ---------------------------------------------------------------------------

typedef __attribute__((ext_vector_type(16))) __bf16 v16bf;
typedef __attribute__((ext_vector_type(8)))  float  v8f;

#define LDS_XSTRIDE 136   // 128 + 8 shorts pad (keeps 16B alignment for b128)

__device__ __forceinline__ unsigned short f2bf_rne(float f) {
  unsigned u = __float_as_uint(f);
  unsigned r = u + 0x7fffu + ((u >> 16) & 1u);   // round-to-nearest-even
  return (unsigned short)(r >> 16);
}
__device__ __forceinline__ float bf2f(unsigned short h) {
  return __uint_as_float(((unsigned)h) << 16);
}

__device__ __forceinline__ void split4_store(const float4 v,
                                             unsigned short* hi_base,
                                             unsigned short* lo_base) {
  unsigned short h0 = f2bf_rne(v.x), h1 = f2bf_rne(v.y);
  unsigned short h2 = f2bf_rne(v.z), h3 = f2bf_rne(v.w);
  unsigned short l0 = f2bf_rne(v.x - bf2f(h0));
  unsigned short l1 = f2bf_rne(v.y - bf2f(h1));
  unsigned short l2 = f2bf_rne(v.z - bf2f(h2));
  unsigned short l3 = f2bf_rne(v.w - bf2f(h3));
  uint2 hv, lv;
  hv.x = (unsigned)h0 | ((unsigned)h1 << 16);
  hv.y = (unsigned)h2 | ((unsigned)h3 << 16);
  lv.x = (unsigned)l0 | ((unsigned)l1 << 16);
  lv.y = (unsigned)l2 | ((unsigned)l3 << 16);
  *(uint2*)hi_base = hv;   // 8B-aligned -> ds_store_b64
  *(uint2*)lo_base = lv;
}

union FragBF {
  v16bf v;
  uint4 q[2];
};

__global__ __launch_bounds__(256)
void proto_dist_kernel(const float* __restrict__ X,
                       const float* __restrict__ P,
                       float* __restrict__ out) {
  extern __shared__ unsigned short smem[];
  // LDS layout (units = unsigned short unless noted):
  unsigned short* XHI = smem;                         // 128 * 136
  unsigned short* XLO = XHI + 128 * LDS_XSTRIDE;      // 128 * 136
  unsigned short* PHI = XLO + 128 * LDS_XSTRIDE;      //  64 * 136
  unsigned short* PLO = PHI +  64 * LDS_XSTRIDE;      //  64 * 136
  float*          XSQ = (float*)(PLO + 64 * LDS_XSTRIDE);  // 128 floats
  float*          PSQ = XSQ + 128;                          //  64 floats

  const int tid = threadIdx.x;
  const size_t m0 = (size_t)blockIdx.x * 128;
  const float4* Xg4 = (const float4*)(X + m0 * 128);  // contiguous [128,128] tile
  const float4* Pg4 = (const float4*)P;

  // ---- Stage X tile as bf16 hi/lo (b128 loads: 512 B/wave/instr, coalesced)
  #pragma unroll 4
  for (int i = 0; i < 16; ++i) {
    int idx = i * 256 + tid;                // float4 index, 0..4095
    float4 x = Xg4[idx];
    int row = idx >> 5;                     // 32 float4 per 128-col row
    int col = (idx & 31) * 4;
    split4_store(x, XHI + row * LDS_XSTRIDE + col, XLO + row * LDS_XSTRIDE + col);
  }
  // ---- Stage prototypes as bf16 hi/lo
  #pragma unroll 4
  for (int i = 0; i < 8; ++i) {
    int idx = i * 256 + tid;                // float4 index, 0..2047
    float4 p = Pg4[idx];
    int row = idx >> 5;
    int col = (idx & 31) * 4;
    split4_store(p, PHI + row * LDS_XSTRIDE + col, PLO + row * LDS_XSTRIDE + col);
  }
  __syncthreads();

  // ---- Deterministic per-row squared norms from hi+lo reconstruction
  if (tid < 128) {
    const unsigned short* h = XHI + tid * LDS_XSTRIDE;
    const unsigned short* l = XLO + tid * LDS_XSTRIDE;
    float s = 0.f;
    for (int c = 0; c < 128; ++c) { float v = bf2f(h[c]) + bf2f(l[c]); s += v * v; }
    XSQ[tid] = s;
  } else if (tid < 192) {
    int k = tid - 128;
    const unsigned short* h = PHI + k * LDS_XSTRIDE;
    const unsigned short* l = PLO + k * LDS_XSTRIDE;
    float s = 0.f;
    for (int c = 0; c < 128; ++c) { float v = bf2f(h[c]) + bf2f(l[c]); s += v * v; }
    PSQ[k] = s;
  }
  __syncthreads();

  // ---- WMMA main loop: wave w handles rows [16w, 16w+16) x 64 prototypes
  const int wave = tid >> 5;      // 0..7 (wave32)
  const int lane = tid & 31;
  const int half = lane >> 4;     // lane group 0-15 / 16-31
  const int lr   = lane & 15;

  v8f acc[4] = {};                // 4 N-tiles of 16x16 f32 accumulators

  #pragma unroll
  for (int ks = 0; ks < 4; ++ks) {       // K-dim (D=128) in steps of 32
    const int d0 = ks * 32;
    // A fragment (16x32 bf16): lane half h holds K=16h..16h+15
    const unsigned short* ab  = XHI + (wave * 16 + lr) * LDS_XSTRIDE + d0 + 16 * half;
    const unsigned short* alb = XLO + (wave * 16 + lr) * LDS_XSTRIDE + d0 + 16 * half;
    FragBF a, al;
    a.q[0]  = *(const uint4*)(ab);
    a.q[1]  = *(const uint4*)(ab + 8);
    al.q[0] = *(const uint4*)(alb);
    al.q[1] = *(const uint4*)(alb + 8);

    #pragma unroll
    for (int n = 0; n < 4; ++n) {
      // B fragment (32x16 bf16): lane = column (prototype), half h holds K=16h..16h+15
      const unsigned short* bb  = PHI + (n * 16 + lr) * LDS_XSTRIDE + d0 + 16 * half;
      const unsigned short* blb = PLO + (n * 16 + lr) * LDS_XSTRIDE + d0 + 16 * half;
      FragBF b, bl;
      b.q[0]  = *(const uint4*)(bb);
      b.q[1]  = *(const uint4*)(bb + 8);
      bl.q[0] = *(const uint4*)(blb);
      bl.q[1] = *(const uint4*)(blb + 8);

      // split-bf16 emulated-fp32 product: hi*hi + hi*lo + lo*hi (fp32 accum)
      acc[n] = __builtin_amdgcn_wmma_f32_16x16x32_bf16(
          false, a.v,  false, b.v,  (short)0, acc[n], false, false);
      acc[n] = __builtin_amdgcn_wmma_f32_16x16x32_bf16(
          false, a.v,  false, bl.v, (short)0, acc[n], false, false);
      acc[n] = __builtin_amdgcn_wmma_f32_16x16x32_bf16(
          false, al.v, false, b.v,  (short)0, acc[n], false, false);
    }
  }

  // ---- Epilogue: dist = xsq + psq - 2*cross
  // C layout: lanes 0-15: VGPR r = C[M=r][N=lane]; lanes 16-31: C[M=r+8][N=lane-16]
  #pragma unroll
  for (int n = 0; n < 4; ++n) {
    float ps = PSQ[16 * n + lr];
    #pragma unroll
    for (int r = 0; r < 8; ++r) {
      int mlocal = wave * 16 + r + 8 * half;
      float d = XSQ[mlocal] + ps - 2.0f * acc[n][r];
      out[(m0 + mlocal) * 64 + 16 * n + lr] = d;
    }
  }

  // ---- Tuple output: append prototypes after the 131072*64 distances
  if (blockIdx.x == 0) {
    float4* out4 = (float4*)(out + (size_t)131072 * 64);
    #pragma unroll 4
    for (int i = 0; i < 8; ++i) {
      int idx = i * 256 + tid;              // float4 index, 0..2047
      out4[idx] = Pg4[idx];
    }
  }
}

extern "C" void kernel_launch(void* const* d_in, const int* in_sizes, int n_in,
                              void* d_out, int out_size, void* d_ws, size_t ws_size,
                              hipStream_t stream) {
  (void)in_sizes; (void)n_in; (void)out_size; (void)d_ws; (void)ws_size;
  const float* X = (const float*)d_in[0];   // [64,2048,128] fp32
  const float* P = (const float*)d_in[1];   // [64,128] fp32
  float* out = (float*)d_out;

  const size_t shmem =
      (size_t)(2 * 128 * LDS_XSTRIDE + 2 * 64 * LDS_XSTRIDE) * sizeof(unsigned short)
      + (128 + 64) * sizeof(float);         // ~105 KB, fits 320 KB/WGP

  proto_dist_kernel<<<1024, 256, shmem, stream>>>(X, P, out);
}
